// LlamaAttention_54571854463276
// MI455X (gfx1250) — matrix operand order repaired and checked by reading, not compiled, not run on hardware
//
#include <hip/hip_runtime.h>
#include <hip/hip_bf16.h>
#include <stdint.h>

// Problem constants (from reference): B=1
#define H_   32
#define KVH_ 8
#define HD_  128
#define D_   4096
#define S_   2048
#define KVD_ (KVH_ * HD_)   // 1024
#define GRP_ (H_ / KVH_)    // 4

typedef __attribute__((ext_vector_type(16))) __bf16 v16bf;
typedef __attribute__((ext_vector_type(8)))  float  v8f;
typedef unsigned short u16t;
typedef unsigned int   u32t;

#define NEGM (-3.402823466e38f)

__device__ __forceinline__ u16t f2bf(float f) {
  u32t u = __float_as_uint(f);
  u32t r = (u + 0x7FFFu + ((u >> 16) & 1u)) >> 16;  // round-to-nearest-even
  return (u16t)r;
}

// ---- WMMA fragment loaders (bf16, 16x16x32, wave32 layouts per CDNA5 ISA) ----
// A (16x32, MxK): lanes 0-15 -> M=lane, K in {0..7,16..23}; lanes 16-31 -> K in {8..15,24..31}
__device__ __forceinline__ v16bf frag_a_bf16(const u16t* A, int lda, int row0, int k0, int lane) {
  int r  = row0 + (lane & 15);
  int kb = k0 + ((lane >> 4) << 3);            // +0 or +8
  const u16t* p = A + (size_t)r * lda + kb;
  union { uint4 u[2]; v16bf v; } f;
  f.u[0] = *(const uint4*)(p);                 // K kb..kb+7
  f.u[1] = *(const uint4*)(p + 16);            // K kb+16..kb+23
  return f.v;
}

// B (32x16, KxN) with Bm[k][n] = W[n][k] (W row-major NxK): lane 0-15 -> N=lane, K in {0..7,16..23};
// lanes 16-31 -> same N, K in {8..15,24..31}.
__device__ __forceinline__ v16bf frag_b_bf16(const u16t* Wm, int ldw, int col0, int k0, int lane) {
  int r  = col0 + (lane & 15);
  int kb = k0 + ((lane >> 4) << 3);            // +0 or +8
  const u16t* p = Wm + (size_t)r * ldw + kb;
  union { uint4 u[2]; v16bf v; } f;
  f.u[0] = *(const uint4*)(p);
  f.u[1] = *(const uint4*)(p + 16);
  return f.v;
}

// A fragment built from f32 source (attn probabilities), converted to bf16 inline
__device__ __forceinline__ v16bf frag_a_f32cvt(const float* A, int lda, int row0, int k0, int lane) {
  int r  = row0 + (lane & 15);
  int kb = k0 + ((lane >> 4) << 3);
  const float* p = A + (size_t)r * lda + kb;
  union { u16t s[16]; v16bf v; } f;
#pragma unroll
  for (int i = 0; i < 8; ++i) f.s[i] = f2bf(p[i]);
#pragma unroll
  for (int i = 0; i < 8; ++i) f.s[8 + i] = f2bf(p[16 + i]);
  return f.v;
}

#define WMMA_BF16(a, b, c) \
  __builtin_amdgcn_wmma_f32_16x16x32_bf16(false, (a), false, (b), (short)0, (c), false, false)

// =================== f32 -> bf16 conversion ===================
__global__ void cvt_bf16(const float* __restrict__ src, u16t* __restrict__ dst, int n) {
  int i = blockIdx.x * blockDim.x + threadIdx.x;
  if (i < n) dst[i] = f2bf(src[i]);
}

// =================== generic C = A @ W^T (bf16 in, f32 out) ===================
// Block 256 thr = 8 waves; tile 128x128; wave tile 32(M) x 64(N); K step 32.
__global__ __launch_bounds__(256) void gemm_xw_f32(
    const u16t* __restrict__ A, const u16t* __restrict__ Wm, float* __restrict__ C,
    int M, int N, int K) {
  const int lane = threadIdx.x & 31;
  const int wave = threadIdx.x >> 5;
  const int m0 = blockIdx.y * 128 + (wave & 3) * 32;
  const int n0 = blockIdx.x * 128 + (wave >> 2) * 64;
  (void)M;
  v8f acc[2][4];
#pragma unroll
  for (int i = 0; i < 2; ++i)
#pragma unroll
    for (int j = 0; j < 4; ++j) acc[i][j] = {};
  for (int k0 = 0; k0 < K; k0 += 32) {
    v16bf a0 = frag_a_bf16(A, K, m0,      k0, lane);
    v16bf a1 = frag_a_bf16(A, K, m0 + 16, k0, lane);
#pragma unroll
    for (int j = 0; j < 4; ++j) {
      v16bf b = frag_b_bf16(Wm, K, n0 + 16 * j, k0, lane);
      acc[0][j] = WMMA_BF16(a0, b, acc[0][j]);
      acc[1][j] = WMMA_BF16(a1, b, acc[1][j]);
    }
  }
  const int colb = lane & 15;
  const int rowb = (lane >> 4) << 3;
#pragma unroll
  for (int i = 0; i < 2; ++i)
#pragma unroll
    for (int j = 0; j < 4; ++j) {
      int col = n0 + 16 * j + colb;
#pragma unroll
      for (int r = 0; r < 8; ++r)
        C[(size_t)(m0 + 16 * i + rowb + r) * N + col] = acc[i][j][r];
    }
}

// =================== RoPE + per-head repack (f32 -> bf16) ===================
// grid: (S, H+KVH), block HD. q_t: [H][S][HD], k_t: [KVH][S][HD]
__global__ void rope_pack(const float* __restrict__ q_lin, const float* __restrict__ k_lin,
                          const float* __restrict__ cosp, const float* __restrict__ sinp,
                          u16t* __restrict__ q_t, u16t* __restrict__ k_t) {
  const int s = blockIdx.x, hh = blockIdx.y, d = threadIdx.x;
  const float c  = cosp[(size_t)s * HD_ + d];
  const float sn = sinp[(size_t)s * HD_ + d];
  if (hh < H_) {
    const float* src = q_lin + (size_t)s * D_ + hh * HD_;
    float x  = src[d];
    float xr = (d < HD_ / 2) ? -src[d + HD_ / 2] : src[d - HD_ / 2];
    q_t[((size_t)hh * S_ + s) * HD_ + d] = f2bf(x * c + xr * sn);
  } else {
    const int h = hh - H_;
    const float* src = k_lin + (size_t)s * KVD_ + h * HD_;
    float x  = src[d];
    float xr = (d < HD_ / 2) ? -src[d + HD_ / 2] : src[d - HD_ / 2];
    k_t[((size_t)h * S_ + s) * HD_ + d] = f2bf(x * c + xr * sn);
  }
}

// =================== V transpose pack: v_t[kv][d][s] = v_lin[s][kv*HD+d] ===================
__global__ void v_pack(const float* __restrict__ v_lin, u16t* __restrict__ v_t) {
  int idx = blockIdx.x * blockDim.x + threadIdx.x;   // total = KVH*HD*S = 2^21
  int s = idx & (S_ - 1);
  int d = (idx >> 11) & (HD_ - 1);
  int h = idx >> 18;
  v_t[idx] = f2bf(v_lin[(size_t)s * KVD_ + h * HD_ + d]);
}

// =================== logits = scale * Q_h K_h^T with causal mask ===================
// grid: (S/128, S/128, H). Tiles fully above the diagonal just get -FLT_MAX fill.
__global__ __launch_bounds__(256) void qk_logits(
    const u16t* __restrict__ q_t, const u16t* __restrict__ k_t, float* __restrict__ P) {
  const int h = blockIdx.z;
  float* Ph = P + (size_t)h * S_ * S_;
  if (blockIdx.x > blockIdx.y) {              // fully masked (n0 > m0+127)
    int m0 = blockIdx.y * 128, n0 = blockIdx.x * 128;
    for (int i = threadIdx.x; i < 128 * 128; i += 256) {
      int r = i >> 7, c = i & 127;
      Ph[(size_t)(m0 + r) * S_ + n0 + c] = NEGM;
    }
    return;
  }
  const u16t* Q  = q_t + (size_t)h * S_ * HD_;
  const u16t* Kh = k_t + (size_t)(h / GRP_) * S_ * HD_;
  const int lane = threadIdx.x & 31;
  const int wave = threadIdx.x >> 5;
  const int m0 = blockIdx.y * 128 + (wave & 3) * 32;
  const int n0 = blockIdx.x * 128 + (wave >> 2) * 64;
  v8f acc[2][4];
#pragma unroll
  for (int i = 0; i < 2; ++i)
#pragma unroll
    for (int j = 0; j < 4; ++j) acc[i][j] = {};
#pragma unroll
  for (int k0 = 0; k0 < HD_; k0 += 32) {
    v16bf a0 = frag_a_bf16(Q, HD_, m0,      k0, lane);
    v16bf a1 = frag_a_bf16(Q, HD_, m0 + 16, k0, lane);
#pragma unroll
    for (int j = 0; j < 4; ++j) {
      v16bf b = frag_b_bf16(Kh, HD_, n0 + 16 * j, k0, lane);
      acc[0][j] = WMMA_BF16(a0, b, acc[0][j]);
      acc[1][j] = WMMA_BF16(a1, b, acc[1][j]);
    }
  }
  const float scaling = 0.08838834764831845f;  // HD^-0.5
  const int colb = lane & 15;
  const int rowb = (lane >> 4) << 3;
#pragma unroll
  for (int i = 0; i < 2; ++i)
#pragma unroll
    for (int j = 0; j < 4; ++j) {
      int col = n0 + 16 * j + colb;
#pragma unroll
      for (int r = 0; r < 8; ++r) {
        int row = m0 + 16 * i + rowb + r;
        float v = acc[i][j][r] * scaling;
        Ph[(size_t)row * S_ + col] = (col <= row) ? v : NEGM;
      }
    }
}

// =================== row softmax (in place) * head_scale ===================
// grid: (S, H), block 256. One read + one write of each row.
__global__ __launch_bounds__(256) void softmax_scale(float* __restrict__ P,
                                                     const float* __restrict__ head_scale) {
  const int h = blockIdx.y, q = blockIdx.x, t = threadIdx.x;
  float* row = P + ((size_t)h * S_ + q) * S_;
  __shared__ float red[256];
  float vals[S_ / 256];
  float m = NEGM;
#pragma unroll
  for (int i = 0; i < S_ / 256; ++i) {
    vals[i] = row[t + i * 256];
    m = fmaxf(m, vals[i]);
  }
  red[t] = m; __syncthreads();
  for (int o = 128; o > 0; o >>= 1) {
    if (t < o) red[t] = fmaxf(red[t], red[t + o]);
    __syncthreads();
  }
  m = red[0]; __syncthreads();
  float s = 0.f;
#pragma unroll
  for (int i = 0; i < S_ / 256; ++i) { vals[i] = __expf(vals[i] - m); s += vals[i]; }
  red[t] = s; __syncthreads();
  for (int o = 128; o > 0; o >>= 1) {
    if (t < o) red[t] += red[t + o];
    __syncthreads();
  }
  const float sc = head_scale[h] / red[0];
#pragma unroll
  for (int i = 0; i < S_ / 256; ++i) row[t + i * 256] = vals[i] * sc;
}

// =================== ctx = P_h @ V_h (f32 probs cvt->bf16 inline; bf16 out) ===================
// grid: (S/128, H). Writes ctx[s][h*HD + d] bf16 for the O projection.
__global__ __launch_bounds__(256) void pv_gemm(
    const float* __restrict__ P, const u16t* __restrict__ v_t, u16t* __restrict__ ctx) {
  const int h = blockIdx.y;
  const float* Ph = P + (size_t)h * S_ * S_;
  const u16t* Vh  = v_t + (size_t)(h / GRP_) * HD_ * S_;   // [d][s], ld = S
  const int lane = threadIdx.x & 31;
  const int wave = threadIdx.x >> 5;
  const int m0 = blockIdx.x * 128 + (wave & 3) * 32;
  const int n0 = (wave >> 2) * 64;                          // N = HD = 128 total
  v8f acc[2][4];
#pragma unroll
  for (int i = 0; i < 2; ++i)
#pragma unroll
    for (int j = 0; j < 4; ++j) acc[i][j] = {};
  for (int k0 = 0; k0 < S_; k0 += 32) {
    v16bf a0 = frag_a_f32cvt(Ph, S_, m0,      k0, lane);
    v16bf a1 = frag_a_f32cvt(Ph, S_, m0 + 16, k0, lane);
#pragma unroll
    for (int j = 0; j < 4; ++j) {
      v16bf b = frag_b_bf16(Vh, S_, n0 + 16 * j, k0, lane);
      acc[0][j] = WMMA_BF16(a0, b, acc[0][j]);
      acc[1][j] = WMMA_BF16(a1, b, acc[1][j]);
    }
  }
  const int colb = lane & 15;
  const int rowb = (lane >> 4) << 3;
#pragma unroll
  for (int i = 0; i < 2; ++i)
#pragma unroll
    for (int j = 0; j < 4; ++j) {
      int col = h * HD_ + n0 + 16 * j + colb;
#pragma unroll
      for (int r = 0; r < 8; ++r)
        ctx[(size_t)(m0 + 16 * i + rowb + r) * D_ + col] = f2bf(acc[i][j][r]);
    }
}

// =========================== launcher ===========================
extern "C" void kernel_launch(void* const* d_in, const int* in_sizes, int n_in,
                              void* d_out, int out_size, void* d_ws, size_t ws_size,
                              hipStream_t stream) {
  (void)in_sizes; (void)n_in; (void)out_size; (void)ws_size;
  const float* hs     = (const float*)d_in[0];
  const float* cosp   = (const float*)d_in[1];
  const float* sinp   = (const float*)d_in[2];
  /* d_in[3] attention_mask: causal mask applied analytically in qk_logits */
  const float* Wq     = (const float*)d_in[4];
  const float* Wk     = (const float*)d_in[5];
  const float* Wv     = (const float*)d_in[6];
  const float* Wo     = (const float*)d_in[7];
  const float* hscale = (const float*)d_in[8];

  float* out_attn = (float*)d_out;                  // (S, D) attn_output
  float* P = out_attn + (size_t)S_ * D_;            // (H, S, S) attn_weights

  char* w = (char*)d_ws;
  auto carve = [&](size_t bytes) -> void* {
    void* p = (void*)w;
    w += (bytes + 255) & ~(size_t)255;
    return p;
  };
  u16t* hs_bf = (u16t*)carve((size_t)S_ * D_ * 2);
  u16t* wq_bf = (u16t*)carve((size_t)D_ * D_ * 2);
  u16t* wk_bf = (u16t*)carve((size_t)KVD_ * D_ * 2);
  u16t* wv_bf = (u16t*)carve((size_t)KVD_ * D_ * 2);
  u16t* wo_bf = (u16t*)carve((size_t)D_ * D_ * 2);
  float* q_lin = (float*)carve((size_t)S_ * D_ * 4);
  float* k_lin = (float*)carve((size_t)S_ * KVD_ * 4);
  float* v_lin = (float*)carve((size_t)S_ * KVD_ * 4);
  u16t* q_t = (u16t*)carve((size_t)H_ * S_ * HD_ * 2);
  u16t* k_t = (u16t*)carve((size_t)KVH_ * S_ * HD_ * 2);
  u16t* v_t = (u16t*)carve((size_t)KVH_ * S_ * HD_ * 2);
  u16t* ctx = (u16t*)carve((size_t)S_ * D_ * 2);

  // 1) f32 -> bf16 operand packs
  cvt_bf16<<<(S_ * D_) / 256, 256, 0, stream>>>(hs, hs_bf, S_ * D_);
  cvt_bf16<<<(D_ * D_) / 256, 256, 0, stream>>>(Wq, wq_bf, D_ * D_);
  cvt_bf16<<<(KVD_ * D_) / 256, 256, 0, stream>>>(Wk, wk_bf, KVD_ * D_);
  cvt_bf16<<<(KVD_ * D_) / 256, 256, 0, stream>>>(Wv, wv_bf, KVD_ * D_);
  cvt_bf16<<<(D_ * D_) / 256, 256, 0, stream>>>(Wo, wo_bf, D_ * D_);

  // 2) Q/K/V projections (WMMA bf16, f32 accumulate)
  gemm_xw_f32<<<dim3(D_ / 128, S_ / 128), 256, 0, stream>>>(hs_bf, wq_bf, q_lin, S_, D_, D_);
  gemm_xw_f32<<<dim3(KVD_ / 128, S_ / 128), 256, 0, stream>>>(hs_bf, wk_bf, k_lin, S_, KVD_, D_);
  gemm_xw_f32<<<dim3(KVD_ / 128, S_ / 128), 256, 0, stream>>>(hs_bf, wv_bf, v_lin, S_, KVD_, D_);

  // 3) RoPE + head-major repack; V transpose
  rope_pack<<<dim3(S_, H_ + KVH_), HD_, 0, stream>>>(q_lin, k_lin, cosp, sinp, q_t, k_t);
  v_pack<<<(KVH_ * HD_ * S_) / 256, 256, 0, stream>>>(v_lin, v_t);

  // 4) causal logits -> attn_weights buffer (raw)
  qk_logits<<<dim3(S_ / 128, S_ / 128, H_), 256, 0, stream>>>(q_t, k_t, P);

  // 5) softmax + head_scale in place (final attn_weights output)
  softmax_scale<<<dim3(S_, H_), 256, 0, stream>>>(P, hscale);

  // 6) ctx = P @ V  (per head)
  pv_gemm<<<dim3(S_ / 128, H_), 256, 0, stream>>>(P, v_t, ctx);

  // 7) attn_output = ctx @ Wo^T
  gemm_xw_f32<<<dim3(D_ / 128, S_ / 128), 256, 0, stream>>>(ctx, wo_bf, out_attn, S_, D_, D_);
}
